// MoEKernel_66331474919709
// MI455X (gfx1250) — matrix operand order repaired and checked by reading, not compiled, not run on hardware
//
#include <hip/hip_runtime.h>
#include <math.h>

#define TT 32768
#define DD 1024
#define HH 2048

typedef __attribute__((ext_vector_type(16))) __bf16 v16bf;
typedef __attribute__((ext_vector_type(8)))  float  v8f;
typedef __attribute__((ext_vector_type(4)))  int    v4i_;

#define LPAD2 72   // bf16 elements per LDS row (64 data + 8 pad) -> 144B pitch

#define AS1 __attribute__((address_space(1)))
#define AS3 __attribute__((address_space(3)))

// Async-to-LDS path availability (gfx1250); guarded so fallback still compiles.
#if defined(__HIP_DEVICE_COMPILE__) && defined(__has_builtin)
#  if __has_builtin(__builtin_amdgcn_global_load_async_to_lds_b128)
#    define ASYNC_OK 1
#  endif
#endif
#ifndef ASYNC_OK
#  define ASYNC_OK 0
#endif

__device__ __forceinline__ void wait_async0() {
#if defined(__has_builtin)
#  if __has_builtin(__builtin_amdgcn_s_wait_asynccnt)
    __builtin_amdgcn_s_wait_asynccnt(0);
#  else
    asm volatile("s_wait_asynccnt 0" ::: "memory");
#  endif
#else
    asm volatile("s_wait_asynccnt 0" ::: "memory");
#endif
}

__device__ __forceinline__ float gelu_f(float v) {
    return 0.5f * v * (1.0f + erff(v * 0.70710678118654752440f));
}
__device__ __forceinline__ float sanitize_f(float v) {
    if (v != v) v = 0.0f;
    return fminf(10.0f, fmaxf(-10.0f, v));
}

// ---------------------------------------------------------------------------
// K0: f32 -> bf16 bulk conversion (x, sh_w1, sh_w2), 8 elements/thread.
// ---------------------------------------------------------------------------
__global__ __launch_bounds__(256)
void cvt_bf16_kernel(const float* __restrict__ src, void* __restrict__ dstv)
{
    __bf16* dst = (__bf16*)dstv;
    const size_t i0 = ((size_t)blockIdx.x * 256 + threadIdx.x) * 8;
    const float4 a = *(const float4*)(src + i0);
    const float4 b = *(const float4*)(src + i0 + 4);
    union { __bf16 h[8]; int4 v; } u;
    u.h[0] = (__bf16)a.x; u.h[1] = (__bf16)a.y; u.h[2] = (__bf16)a.z; u.h[3] = (__bf16)a.w;
    u.h[4] = (__bf16)b.x; u.h[5] = (__bf16)b.y; u.h[6] = (__bf16)b.z; u.h[7] = (__bf16)b.w;
    *(int4*)(dst + i0) = u.v;
}

// ---------------------------------------------------------------------------
// K1: router (deterministic, no atomics)
// ---------------------------------------------------------------------------
__global__ __launch_bounds__(256)
void router_kernel(const float* __restrict__ x, const float* __restrict__ rw,
                   const float* __restrict__ ebias,
                   float* __restrict__ logits, float* __restrict__ combine,
                   float* __restrict__ partials)
{
    const int lane  = threadIdx.x & 31;
    const int wave  = threadIdx.x >> 5;
    const int entry = blockIdx.x * 8 + wave;

    float zsum = 0.f;
    float u0 = 0.f, u1 = 0.f, u2 = 0.f, u3 = 0.f;
    float mx[4] = {-1e30f, -1e30f, -1e30f, -1e30f};
    float se[4] = {0.f, 0.f, 0.f, 0.f};

    const float b0 = ebias[0], b1 = ebias[1], b2 = ebias[2], b3 = ebias[3];

    for (int i = 0; i < 8; ++i) {
        const int t = entry * 8 + i;
        const float* xr = x + (size_t)t * DD;
        float a0 = 0.f, a1 = 0.f, a2 = 0.f, a3 = 0.f;
        for (int j = lane; j < DD; j += 32) {
            const float xv = xr[j];
            a0 += xv * rw[j];
            a1 += xv * rw[DD + j];
            a2 += xv * rw[2 * DD + j];
            a3 += xv * rw[3 * DD + j];
        }
#pragma unroll
        for (int off = 16; off; off >>= 1) {
            a0 += __shfl_xor(a0, off, 32);
            a1 += __shfl_xor(a1, off, 32);
            a2 += __shfl_xor(a2, off, 32);
            a3 += __shfl_xor(a3, off, 32);
        }
        const float l0 = a0 + b0, l1 = a1 + b1, l2 = a2 + b2, l3 = a3 + b3;
        const float m  = fmaxf(fmaxf(l0, l1), fmaxf(l2, l3));
        const float e0 = expf(l0 - m), e1 = expf(l1 - m);
        const float e2 = expf(l2 - m), e3 = expf(l3 - m);
        const float Zt  = e0 + e1 + e2 + e3;
        const float iZt = 1.0f / Zt;
        if (lane == 0) {
            float4 lg; lg.x = l0; lg.y = l1; lg.z = l2; lg.w = l3;
            *(float4*)(logits + 4 * (size_t)t) = lg;
            float4 cb; cb.x = e0 * iZt; cb.y = e1 * iZt; cb.z = e2 * iZt; cb.w = e3 * iZt;
            *(float4*)(combine + 4 * (size_t)t) = cb;
        }
        u0 += e0 * iZt; u1 += e1 * iZt; u2 += e2 * iZt; u3 += e3 * iZt;
        const float lse = m + logf(Zt);
        const float zm  = fminf(lse, 10.0f);
        zsum += zm * zm;
        const float l[4] = {l0, l1, l2, l3};
#pragma unroll
        for (int s = 0; s < 4; ++s) {
            if (l[s] > mx[s]) { se[s] = se[s] * expf(mx[s] - l[s]) + 1.0f; mx[s] = l[s]; }
            else              { se[s] += expf(l[s] - mx[s]); }
        }
    }
    if (lane == 0) {
        float* p = partials + (size_t)entry * 16;
        p[0] = zsum;
        p[1] = u0; p[2] = u1; p[3] = u2; p[4] = u3;
        p[5] = mx[0]; p[6] = mx[1]; p[7] = mx[2]; p[8] = mx[3];
        p[9] = se[0]; p[10] = se[1]; p[11] = se[2]; p[12] = se[3];
    }
}

// ---------------------------------------------------------------------------
// K2: merge partials; losses; dispatch stats
// ---------------------------------------------------------------------------
__global__ __launch_bounds__(256)
void finalize_kernel(const float* __restrict__ partials,
                     const float* __restrict__ rw,
                     float* __restrict__ stats, float* __restrict__ loss_out)
{
    __shared__ float sZ[256];
    __shared__ float sU[4][256];
    __shared__ float sMx[4][256];
    __shared__ float sSe[4][256];
    __shared__ float sG[16];
    const int tid = threadIdx.x;

    float z = 0.f, u[4] = {0.f, 0.f, 0.f, 0.f};
    float mx[4] = {-1e30f, -1e30f, -1e30f, -1e30f};
    float se[4] = {0.f, 0.f, 0.f, 0.f};
    for (int e = tid; e < 4096; e += 256) {
        const float* p = partials + (size_t)e * 16;
        z += p[0];
#pragma unroll
        for (int s = 0; s < 4; ++s) u[s] += p[1 + s];
#pragma unroll
        for (int s = 0; s < 4; ++s) {
            const float m2 = p[5 + s], s2 = p[9 + s];
            const float M  = fmaxf(mx[s], m2);
            se[s] = se[s] * expf(mx[s] - M) + s2 * expf(m2 - M);
            mx[s] = M;
        }
    }
    sZ[tid] = z;
#pragma unroll
    for (int s = 0; s < 4; ++s) { sU[s][tid] = u[s]; sMx[s][tid] = mx[s]; sSe[s][tid] = se[s]; }
    __syncthreads();
    for (int off = 128; off; off >>= 1) {
        if (tid < off) {
            sZ[tid] += sZ[tid + off];
#pragma unroll
            for (int s = 0; s < 4; ++s) {
                sU[s][tid] += sU[s][tid + off];
                const float m1 = sMx[s][tid], m2 = sMx[s][tid + off];
                const float M  = fmaxf(m1, m2);
                sSe[s][tid] = sSe[s][tid] * expf(m1 - M) + sSe[s][tid + off] * expf(m2 - M);
                sMx[s][tid] = M;
            }
        }
        __syncthreads();
    }
    if (tid < 16) {
        const int i = tid >> 2, j = tid & 3;
        float d = 0.f;
        for (int k = 0; k < DD; ++k) d += rw[i * DD + k] * rw[j * DD + k];
        sG[tid] = d;
    }
    __syncthreads();
    if (tid == 0) {
        float nrm[4];
#pragma unroll
        for (int i = 0; i < 4; ++i) nrm[i] = sqrtf(sG[i * 4 + i]) + 1e-8f;
        float ortho = 0.f;
        for (int i = 0; i < 4; ++i)
            for (int j = 0; j < 4; ++j) {
                const float g   = sG[i * 4 + j] / (nrm[i] * nrm[j]);
                const float off = g - ((i == j) ? 1.0f : 0.0f);
                ortho += off * off;
            }
        ortho *= (1.0f / 12.0f);
        const float zloss = sZ[0] / (float)TT;
        float rloss = 0.f;
        for (int e = 0; e < 4; ++e) {
            const float du = sU[e][0] / (float)TT - 0.25f;
            rloss += du * du;
        }
        rloss *= 4.0f;
        loss_out[0] = rloss + 0.01f * zloss + 0.01f * ortho;
#pragma unroll
        for (int s = 0; s < 4; ++s) { stats[s] = sMx[s][0]; stats[4 + s] = 1.0f / sSe[s][0]; }
    }
}

// ---------------------------------------------------------------------------
// K3: slot_in partials + reduce
// ---------------------------------------------------------------------------
__global__ __launch_bounds__(256)
void slotpart_kernel(const float* __restrict__ x, const float* __restrict__ logits,
                     const float* __restrict__ stats, float* __restrict__ part)
{
    const int b   = blockIdx.x;
    const int tid = threadIdx.x;
    const float m0 = stats[0], m1 = stats[1], m2 = stats[2], m3 = stats[3];
    const float i0 = stats[4], i1 = stats[5], i2 = stats[6], i3 = stats[7];
    float acc[4][4];
#pragma unroll
    for (int s = 0; s < 4; ++s)
#pragma unroll
        for (int c = 0; c < 4; ++c) acc[s][c] = 0.f;
    const int d0 = tid * 4;
    for (int i = 0; i < 256; ++i) {
        const int t = b * 256 + i;
        const float4 lg = *(const float4*)(logits + 4 * (size_t)t);
        const float w0 = expf(lg.x - m0) * i0;
        const float w1 = expf(lg.y - m1) * i1;
        const float w2 = expf(lg.z - m2) * i2;
        const float w3 = expf(lg.w - m3) * i3;
        const float4 xv = *(const float4*)(x + (size_t)t * DD + d0);
        acc[0][0] += w0 * xv.x; acc[0][1] += w0 * xv.y; acc[0][2] += w0 * xv.z; acc[0][3] += w0 * xv.w;
        acc[1][0] += w1 * xv.x; acc[1][1] += w1 * xv.y; acc[1][2] += w1 * xv.z; acc[1][3] += w1 * xv.w;
        acc[2][0] += w2 * xv.x; acc[2][1] += w2 * xv.y; acc[2][2] += w2 * xv.z; acc[2][3] += w2 * xv.w;
        acc[3][0] += w3 * xv.x; acc[3][1] += w3 * xv.y; acc[3][2] += w3 * xv.z; acc[3][3] += w3 * xv.w;
    }
#pragma unroll
    for (int s = 0; s < 4; ++s) {
        float4 o; o.x = acc[s][0]; o.y = acc[s][1]; o.z = acc[s][2]; o.w = acc[s][3];
        *(float4*)(part + (size_t)b * 4096 + s * 1024 + d0) = o;
    }
}

__global__ __launch_bounds__(256)
void slotreduce_kernel(const float* __restrict__ part, float* __restrict__ slot_in)
{
    for (int o = threadIdx.x; o < 4096; o += 256) {
        float s = 0.f;
        for (int b = 0; b < 128; ++b) s += part[(size_t)b * 4096 + o];
        slot_in[o] = s;
    }
}

// ---------------------------------------------------------------------------
// K4: four tiny expert MLPs
// ---------------------------------------------------------------------------
__global__ __launch_bounds__(256)
void experts_kernel(const float* __restrict__ slot_in,
    const float* m_w1, const float* m_b1, const float* m_w2, const float* m_b2,
    const float* m_w3, const float* m_b3,
    const float* l_g,  const float* l_be, const float* l_w1, const float* l_b1,
    const float* l_w2, const float* l_b2,
    const float* c_w1, const float* c_b1, const float* c_w2, const float* c_b2,
    const float* s_w1, const float* s_b1, const float* s_w2, const float* s_b2,
    float* __restrict__ slot_out)
{
    __shared__ float xbuf[1024];
    __shared__ float hbuf[4096];
    __shared__ float h2buf[2048];
    __shared__ float sred[256];
    const int e = blockIdx.x, tid = threadIdx.x;
    for (int i = tid; i < 1024; i += 256) xbuf[i] = slot_in[e * 1024 + i];
    __syncthreads();

    if (e == 0) {
        for (int o = tid; o < 4096; o += 256) {
            const float* wr = m_w1 + (size_t)o * 1024;
            float s = m_b1[o];
            for (int d = 0; d < 1024; ++d) s += xbuf[d] * wr[d];
            hbuf[o] = gelu_f(s);
        }
        __syncthreads();
        for (int o = tid; o < 2048; o += 256) {
            const float* wr = m_w2 + (size_t)o * 4096;
            float s = m_b2[o];
            for (int d = 0; d < 4096; ++d) s += hbuf[d] * wr[d];
            h2buf[o] = gelu_f(s);
        }
        __syncthreads();
        for (int o = tid; o < 1024; o += 256) {
            const float* wr = m_w3 + (size_t)o * 2048;
            float s = m_b3[o];
            for (int d = 0; d < 2048; ++d) s += h2buf[d] * wr[d];
            slot_out[o] = sanitize_f(s);
        }
    } else {
        const float *w1, *bb1, *w2, *bb2;
        if (e == 1)      { w1 = l_w1; bb1 = l_b1; w2 = l_w2; bb2 = l_b2; }
        else if (e == 2) { w1 = c_w1; bb1 = c_b1; w2 = c_w2; bb2 = c_b2; }
        else             { w1 = s_w1; bb1 = s_b1; w2 = s_w2; bb2 = s_b2; }

        if (e == 1) {
            float s = 0.f;
            for (int i = tid; i < 1024; i += 256) s += xbuf[i];
            sred[tid] = s; __syncthreads();
            for (int off = 128; off; off >>= 1) { if (tid < off) sred[tid] += sred[tid + off]; __syncthreads(); }
            const float mu = sred[0] * (1.0f / 1024.0f);
            __syncthreads();
            float v = 0.f;
            for (int i = tid; i < 1024; i += 256) { const float d = xbuf[i] - mu; v += d * d; }
            sred[tid] = v; __syncthreads();
            for (int off = 128; off; off >>= 1) { if (tid < off) sred[tid] += sred[tid + off]; __syncthreads(); }
            const float rstd = rsqrtf(sred[0] * (1.0f / 1024.0f) + 1e-5f);
            for (int i = tid; i < 1024; i += 256)
                hbuf[i] = (xbuf[i] - mu) * rstd * l_g[i] + l_be[i];
            __syncthreads();
        } else {
            for (int i = tid; i < 1024; i += 256) hbuf[i] = xbuf[i];
            __syncthreads();
        }
        for (int o = tid; o < 2048; o += 256) {
            const float* wr = w1 + (size_t)o * 1024;
            float s = bb1[o];
            for (int d = 0; d < 1024; ++d) s += hbuf[d] * wr[d];
            if (e == 1)      s = gelu_f(s);
            else if (e == 2) s = s / (1.0f + expf(-s));
            else             s = tanhf(s);
            h2buf[o] = s;
        }
        __syncthreads();
        for (int o = tid; o < 1024; o += 256) {
            const float* wr = w2 + (size_t)o * 2048;
            float s = bb2[o];
            for (int d = 0; d < 2048; ++d) s += h2buf[d] * wr[d];
            slot_out[e * 1024 + o] = sanitize_f(s);
        }
    }
}

// ---------------------------------------------------------------------------
// WMMA GEMM helpers
// ---------------------------------------------------------------------------
// A fragment (16x32, MxK): lane M = lane&15; K chunks {0..7,16..23}/{8..15,24..31}
__device__ __forceinline__ v16bf frag_a(const __bf16* sm, int row, int kk, int lane) {
    const int r  = row + (lane & 15);
    const int c0 = kk + ((lane & 16) ? 8 : 0);
    const __bf16* p = sm + r * LPAD2 + c0;
    union { int4 v; __bf16 h[8]; } u0, u1;
    u0.v = *(const int4*)(p);
    u1.v = *(const int4*)(p + 16);
    v16bf o;
#pragma unroll
    for (int i = 0; i < 8; ++i) { o[i] = u0.h[i]; o[i + 8] = u1.h[i]; }
    return o;
}
// B fragment (32x16, KxN): lane N = lane&15; K chunks {0..7,16..23}/{8..15,24..31}
__device__ __forceinline__ v16bf frag_b(const __bf16* sm, int row, int kk, int lane) {
    const int r  = row + (lane & 15);
    const int c0 = kk + ((lane & 16) ? 8 : 0);
    const __bf16* p = sm + r * LPAD2 + c0;
    union { int4 v; __bf16 h[8]; } u0, u1;
    u0.v = *(const int4*)(p);
    u1.v = *(const int4*)(p + 16);
    v16bf o;
#pragma unroll
    for (int i = 0; i < 8; ++i) { o[i] = u0.h[i]; o[i + 8] = u1.h[i]; }
    return o;
}

#if ASYNC_OK
// 128x64 bf16 tile: 1024 16B-chunks, 4 per thread, copied global->LDS by the
// async engine (tracked with ASYNCcnt, no VGPR staging).
// Builtin signature (from clang diagnostic): (AS1 v4i*, AS3 v4i*, imm, imm).
__device__ __forceinline__ void async_tile(const __bf16* g, int ld, __bf16* s, int tid) {
#pragma unroll
    for (int k = 0; k < 4; ++k) {
        const int c   = tid + k * 256;
        const int row = c >> 3;
        const int col = (c & 7) * 8;
        __builtin_amdgcn_global_load_async_to_lds_b128(
            (AS1 v4i_*)(g + (size_t)row * ld + col),
            (AS3 v4i_*)(s + row * LPAD2 + col), 0, 0);
    }
}
#endif
__device__ __forceinline__ void sync_tile(const __bf16* g, int ld, __bf16* s, int tid) {
#pragma unroll
    for (int k = 0; k < 4; ++k) {
        const int c   = tid + k * 256;
        const int row = c >> 3;
        const int col = (c & 7) * 8;
        *(int4*)(s + row * LPAD2 + col) = *(const int4*)(g + (size_t)row * ld + col);
    }
}

// ---------------------------------------------------------------------------
// K5: H = gelu(x_bf16 @ w1_bf16^T + sh_b1) -> bf16.  M=TT, N=HH, K=DD.
// 128x128 block, BK=64, double-buffered async LDS pipeline, 16 wmma/iter.
// ---------------------------------------------------------------------------
__global__ __launch_bounds__(256)
void gemm1_kernel(const void* __restrict__ Axv, const void* __restrict__ Bxv,
                  const float* __restrict__ bias, void* __restrict__ Hout)
{
    const __bf16* Ax = (const __bf16*)Axv;
    const __bf16* Bx = (const __bf16*)Bxv;
    __bf16* H = (__bf16*)Hout;
    const int K = DD, N = HH;
    __shared__ __align__(16) __bf16 Asm[2][128 * LPAD2];
    __shared__ __align__(16) __bf16 Bsm[2][128 * LPAD2];

    const int tid = threadIdx.x;
    const int lane = tid & 31, wave = tid >> 5;
    const int wm = (wave >> 1) * 32;
    const int wn = (wave & 1) * 64;
    const int bm = blockIdx.y * 128;
    const int bn = blockIdx.x * 128;

    const __bf16* Ag = Ax + (size_t)bm * K;
    const __bf16* Bg = Bx + (size_t)bn * K;

    v8f acc[2][4];
#pragma unroll
    for (int mi = 0; mi < 2; ++mi)
#pragma unroll
        for (int ni = 0; ni < 4; ++ni)
#pragma unroll
            for (int i = 0; i < 8; ++i) acc[mi][ni][i] = 0.0f;

    const int nk = K / 64;
#if ASYNC_OK
    async_tile(Ag, K, Asm[0], tid);
    async_tile(Bg, K, Bsm[0], tid);
#else
    sync_tile(Ag, K, Asm[0], tid);
    sync_tile(Bg, K, Bsm[0], tid);
#endif
    for (int kt = 0; kt < nk; ++kt) {
        const int cur = kt & 1;
#if ASYNC_OK
        wait_async0();                               // stage kt copies (this wave) done
#endif
        __syncthreads();                             // all waves' copies visible
#if ASYNC_OK
        if (kt + 1 < nk) {                           // overlap next-stage copies with wmma
            async_tile(Ag + (kt + 1) * 64, K, Asm[1 - cur], tid);
            async_tile(Bg + (kt + 1) * 64, K, Bsm[1 - cur], tid);
        }
#endif
#pragma unroll
        for (int kk = 0; kk < 64; kk += 32) {
            v16bf af[2];
            af[0] = frag_a(Asm[cur], wm, kk, lane);
            af[1] = frag_a(Asm[cur], wm + 16, kk, lane);
            v16bf bfr[4];
#pragma unroll
            for (int ni = 0; ni < 4; ++ni) bfr[ni] = frag_b(Bsm[cur], wn + ni * 16, kk, lane);
#pragma unroll
            for (int mi = 0; mi < 2; ++mi)
#pragma unroll
                for (int ni = 0; ni < 4; ++ni)
                    acc[mi][ni] = __builtin_amdgcn_wmma_f32_16x16x32_bf16(
                        false, af[mi], false, bfr[ni], (short)0, acc[mi][ni], false, false);
        }
        __syncthreads();                             // reads of buf[cur] done
#if !ASYNC_OK
        if (kt + 1 < nk) {
            sync_tile(Ag + (kt + 1) * 64, K, Asm[1 - cur], tid);
            sync_tile(Bg + (kt + 1) * 64, K, Bsm[1 - cur], tid);
        }
#endif
    }

    const int rbase = (lane >> 4) << 3;
    const int ncl   = lane & 15;
#pragma unroll
    for (int ni = 0; ni < 4; ++ni) {
        const int n = bn + wn + ni * 16 + ncl;
        const float bv = bias[n];
#pragma unroll
        for (int mi = 0; mi < 2; ++mi) {
#pragma unroll
            for (int i = 0; i < 8; ++i) {
                const int m = bm + wm + mi * 16 + rbase + i;
                float v = acc[mi][ni][i] + bv;
                v = gelu_f(v);
                H[(size_t)m * N + n] = (__bf16)v;
            }
        }
    }
}

// ---------------------------------------------------------------------------
// K6: out = H @ w2_bf16^T + sh_b2 + combine @ slot_out.  M=TT, N=DD, K=HH.
// ---------------------------------------------------------------------------
__global__ __launch_bounds__(256)
void gemm2_kernel(const void* __restrict__ Hin, const void* __restrict__ Bxv,
                  const float* __restrict__ bias, const float* __restrict__ combine,
                  const float* __restrict__ slot_out, float* __restrict__ out)
{
    const __bf16* Hb = (const __bf16*)Hin;
    const __bf16* Bx = (const __bf16*)Bxv;
    const int K = HH, N = DD;
    __shared__ __align__(16) __bf16 Asm[2][128 * LPAD2];
    __shared__ __align__(16) __bf16 Bsm[2][128 * LPAD2];

    const int tid = threadIdx.x;
    const int lane = tid & 31, wave = tid >> 5;
    const int wm = (wave >> 1) * 32;
    const int wn = (wave & 1) * 64;
    const int bm = blockIdx.y * 128;
    const int bn = blockIdx.x * 128;

    const __bf16* Ag = Hb + (size_t)bm * K;
    const __bf16* Bg = Bx + (size_t)bn * K;

    v8f acc[2][4];
#pragma unroll
    for (int mi = 0; mi < 2; ++mi)
#pragma unroll
        for (int ni = 0; ni < 4; ++ni)
#pragma unroll
            for (int i = 0; i < 8; ++i) acc[mi][ni][i] = 0.0f;

    const int nk = K / 64;
#if ASYNC_OK
    async_tile(Ag, K, Asm[0], tid);
    async_tile(Bg, K, Bsm[0], tid);
#else
    sync_tile(Ag, K, Asm[0], tid);
    sync_tile(Bg, K, Bsm[0], tid);
#endif
    for (int kt = 0; kt < nk; ++kt) {
        const int cur = kt & 1;
#if ASYNC_OK
        wait_async0();
#endif
        __syncthreads();
#if ASYNC_OK
        if (kt + 1 < nk) {
            async_tile(Ag + (kt + 1) * 64, K, Asm[1 - cur], tid);
            async_tile(Bg + (kt + 1) * 64, K, Bsm[1 - cur], tid);
        }
#endif
#pragma unroll
        for (int kk = 0; kk < 64; kk += 32) {
            v16bf af[2];
            af[0] = frag_a(Asm[cur], wm, kk, lane);
            af[1] = frag_a(Asm[cur], wm + 16, kk, lane);
            v16bf bfr[4];
#pragma unroll
            for (int ni = 0; ni < 4; ++ni) bfr[ni] = frag_b(Bsm[cur], wn + ni * 16, kk, lane);
#pragma unroll
            for (int mi = 0; mi < 2; ++mi)
#pragma unroll
                for (int ni = 0; ni < 4; ++ni)
                    acc[mi][ni] = __builtin_amdgcn_wmma_f32_16x16x32_bf16(
                        false, af[mi], false, bfr[ni], (short)0, acc[mi][ni], false, false);
        }
        __syncthreads();
#if !ASYNC_OK
        if (kt + 1 < nk) {
            sync_tile(Ag + (kt + 1) * 64, K, Asm[1 - cur], tid);
            sync_tile(Bg + (kt + 1) * 64, K, Bsm[1 - cur], tid);
        }
#endif
    }

    const int rbase = (lane >> 4) << 3;
    const int ncl   = lane & 15;
#pragma unroll
    for (int ni = 0; ni < 4; ++ni) {
        const int n = bn + wn + ni * 16 + ncl;
        const float bv = bias[n];
        const float s0 = slot_out[n];
        const float s1 = slot_out[1024 + n];
        const float s2 = slot_out[2048 + n];
        const float s3 = slot_out[3072 + n];
#pragma unroll
        for (int mi = 0; mi < 2; ++mi) {
#pragma unroll
            for (int i = 0; i < 8; ++i) {
                const int m = bm + wm + mi * 16 + rbase + i;
                const float4 cb = *(const float4*)(combine + 4 * (size_t)m);
                out[(size_t)m * N + n] =
                    acc[mi][ni][i] + bv + cb.x * s0 + cb.y * s1 + cb.z * s2 + cb.w * s3;
            }
        }
    }
}

// ---------------------------------------------------------------------------
// Host launcher
// ---------------------------------------------------------------------------
extern "C" void kernel_launch(void* const* d_in, const int* in_sizes, int n_in,
                              void* d_out, int out_size, void* d_ws, size_t ws_size,
                              hipStream_t stream)
{
    (void)in_sizes; (void)n_in; (void)out_size; (void)ws_size;
    const float* x     = (const float*)d_in[0];
    const float* rw    = (const float*)d_in[1];
    const float* ebias = (const float*)d_in[2];
    const float* m_w1  = (const float*)d_in[3];
    const float* m_b1  = (const float*)d_in[4];
    const float* m_w2  = (const float*)d_in[5];
    const float* m_b2  = (const float*)d_in[6];
    const float* m_w3  = (const float*)d_in[7];
    const float* m_b3  = (const float*)d_in[8];
    const float* l_g   = (const float*)d_in[9];
    const float* l_be  = (const float*)d_in[10];
    const float* l_w1  = (const float*)d_in[11];
    const float* l_b1  = (const float*)d_in[12];
    const float* l_w2  = (const float*)d_in[13];
    const float* l_b2  = (const float*)d_in[14];
    const float* c_w1  = (const float*)d_in[15];
    const float* c_b1  = (const float*)d_in[16];
    const float* c_w2  = (const float*)d_in[17];
    const float* c_b2  = (const float*)d_in[18];
    const float* s_w1  = (const float*)d_in[19];
    const float* s_b1  = (const float*)d_in[20];
    const float* s_w2  = (const float*)d_in[21];
    const float* s_b2  = (const float*)d_in[22];
    const float* sh_w1 = (const float*)d_in[23];
    const float* sh_b1 = (const float*)d_in[24];
    const float* sh_w2 = (const float*)d_in[25];
    const float* sh_b2 = (const float*)d_in[26];

    float* out = (float*)d_out;

    // workspace layout
    float* wsf      = (float*)d_ws;
    float* logits   = wsf;                 // 131072 f
    float* combine  = wsf + 131072;        // 131072 f
    float* partials = wsf + 262144;        // 65536 f
    float* stats    = wsf + 327680;        // 64 f
    float* slotpart = wsf + 327744;        // 524288 f
    float* slot_in  = wsf + 852032;        // 4096 f
    float* slot_out = wsf + 856128;        // 4096 f
    char*  wsb      = (char*)d_ws;
    const size_t B_HBUF = (size_t)860224 * 4;                  // TT*HH bf16
    const size_t B_XBF  = B_HBUF + (size_t)TT * HH * 2;        // TT*DD bf16
    const size_t B_W1BF = B_XBF + (size_t)TT * DD * 2;         // HH*DD bf16
    const size_t B_W2BF = B_W1BF + (size_t)HH * DD * 2;        // DD*HH bf16
    void* Hbuf = (void*)(wsb + B_HBUF);
    void* xbf  = (void*)(wsb + B_XBF);
    void* w1bf = (void*)(wsb + B_W1BF);
    void* w2bf = (void*)(wsb + B_W2BF);

    // bulk f32 -> bf16 conversions (x fits in L2 as bf16 -> cheap re-reads)
    cvt_bf16_kernel<<<(TT * (size_t)DD) / 2048, 256, 0, stream>>>(x, xbf);
    cvt_bf16_kernel<<<((size_t)HH * DD) / 2048, 256, 0, stream>>>(sh_w1, w1bf);
    cvt_bf16_kernel<<<((size_t)DD * HH) / 2048, 256, 0, stream>>>(sh_w2, w2bf);

    router_kernel<<<512, 256, 0, stream>>>(x, rw, ebias, logits, combine, partials);
    finalize_kernel<<<1, 256, 0, stream>>>(partials, rw, stats, out + (size_t)TT * DD);
    slotpart_kernel<<<128, 256, 0, stream>>>(x, logits, stats, slotpart);
    slotreduce_kernel<<<1, 256, 0, stream>>>(slotpart, slot_in);
    experts_kernel<<<4, 256, 0, stream>>>(slot_in,
        m_w1, m_b1, m_w2, m_b2, m_w3, m_b3,
        l_g, l_be, l_w1, l_b1, l_w2, l_b2,
        c_w1, c_b1, c_w2, c_b2,
        s_w1, s_b1, s_w2, s_b2, slot_out);

    gemm1_kernel<<<dim3(HH / 128, TT / 128), 256, 0, stream>>>(xbf, w1bf, sh_b1, Hbuf);
    gemm2_kernel<<<dim3(DD / 128, TT / 128), 256, 0, stream>>>(Hbuf, w2bf, sh_b2,
                                                               combine, slot_out, out);
}
